// BinarizedLeNet5Cifar10_26972394619331
// MI455X (gfx1250) — 8-bit matrix operand order repaired and checked by reading, compiled, not run on hardware
//
#include <hip/hip_runtime.h>
#include <math.h>
#include <stdint.h>


typedef int v8i __attribute__((ext_vector_type(8)));

#define EPSBN 1e-5f

__device__ __forceinline__ int8_t sgn8(float v) {
    return (int8_t)(v > 0.f ? 1 : (v < 0.f ? -1 : 0));
}

__device__ __forceinline__ int a_koff(int v, int lane) {
    return ((v & 1) << 2) + (((v >> 1) & 1) << 4) + ((v >> 2) << 5) + ((lane & 16) ? 8 : 0);
}
__device__ __forceinline__ int b_koff(int v, int lane) {
    return ((v & 1) << 2) + (((v >> 1) & 1) << 4) + ((v >> 2) << 5) + ((lane & 16) ? 8 : 0);
}

__global__ void k_prep(const float* __restrict__ w2, const float* __restrict__ wf1,
                       const float* __restrict__ wf2, int8_t* __restrict__ Bw2,
                       int8_t* __restrict__ Bfc1, int8_t* __restrict__ Bfc2) {
    const int total = 64 * 320 + 512 * 4096 + 256 * 512;
    for (int i = blockIdx.x * blockDim.x + threadIdx.x; i < total;
         i += gridDim.x * blockDim.x) {
        if (i < 64 * 320) {
            int n = i / 320, k = i - n * 320;
            int8_t v = 0;
            if (k < 288) v = sgn8(w2[n * 288 + k]);
            Bw2[i] = v;
        } else if (i < 64 * 320 + 512 * 4096) {
            int j = i - 64 * 320;
            Bfc1[j] = sgn8(wf1[j]);
        } else {
            int j = i - 64 * 320 - 512 * 4096;
            Bfc2[j] = sgn8(wf2[j]);
        }
    }
}

__global__ __launch_bounds__(256) void k_conv1(
    const float* __restrict__ X, const float* __restrict__ W,
    const float* __restrict__ bconv, const float* __restrict__ g,
    const float* __restrict__ bb, const float* __restrict__ bm,
    const float* __restrict__ bv, int8_t* __restrict__ s1) {
    __shared__ float xt[3 * 34 * 34];
    const int b = blockIdx.x >> 5;
    const int c = blockIdx.x & 31;
    const int tid = threadIdx.x;
    for (int p = tid; p < 3 * 34 * 34; p += 256) {
        int ci = p / 1156;
        int rem = p - ci * 1156;
        int iy = rem / 34 - 1;
        int ix = rem % 34 - 1;
        float v = 0.f;
        if (iy >= 0 && iy < 32 && ix >= 0 && ix < 32)
            v = X[((b * 3 + ci) * 32 + iy) * 32 + ix];
        xt[p] = v;
    }
    float ws[27];
#pragma unroll
    for (int j = 0; j < 27; ++j) {
        float w = W[c * 27 + j];
        ws[j] = (w > 0.f) ? 1.f : (w < 0.f ? -1.f : 0.f);
    }
    __syncthreads();
    const int px = tid & 15, py = tid >> 4;
    const float bi = bconv[c];
    float mx = -1e30f;
#pragma unroll
    for (int sub = 0; sub < 4; ++sub) {
        int yy = 2 * py + (sub >> 1);
        int xx = 2 * px + (sub & 1);
        float acc = bi;
#pragma unroll
        for (int ci = 0; ci < 3; ++ci)
#pragma unroll
            for (int dy = 0; dy < 3; ++dy)
#pragma unroll
                for (int dx = 0; dx < 3; ++dx)
                    acc += xt[ci * 1156 + (yy + dy) * 34 + (xx + dx)] *
                           ws[ci * 9 + dy * 3 + dx];
        mx = fmaxf(mx, acc);
    }
    float inv = g[c] / sqrtf(bv[c] + EPSBN);
    float h = mx * inv + (bb[c] - bm[c] * inv);
    h = fminf(1.f, fmaxf(-1.f, h));
    s1[((b * 32 + c) * 16 + py) * 16 + px] = sgn8(h);
}

__global__ __launch_bounds__(128) void k_conv2(
    const int8_t* __restrict__ s1, const int8_t* __restrict__ Bw,
    const float* __restrict__ bconv, const float* __restrict__ g,
    const float* __restrict__ bb, const float* __restrict__ bm,
    const float* __restrict__ bv, int8_t* __restrict__ s2) {
    __shared__ int8_t Alds[16 * 320];
    const int b = blockIdx.x >> 4;
    const int pl0 = (blockIdx.x & 15) << 2;
    const int tid = threadIdx.x;
    for (int p = tid; p < 16 * 320; p += 128) {
        int mrow = p / 320;
        int k = p - mrow * 320;
        int8_t val = 0;
        if (k < 288) {
            int pl = pl0 + (mrow >> 2);
            int sub = mrow & 3;
            int yy = 2 * (pl >> 3) + (sub >> 1);
            int xx = 2 * (pl & 7) + (sub & 1);
            int ci = k / 9;
            int rs = k - ci * 9;
            int iy = yy + rs / 3 - 1;
            int ix = xx + rs % 3 - 1;
            if (iy >= 0 && iy < 16 && ix >= 0 && ix < 16)
                val = s1[((b * 32 + ci) * 16 + iy) * 16 + ix];
        }
        Alds[p] = val;
    }
    __syncthreads();
    const int lane = tid & 31;
    const int wv = tid >> 5;
    const int mloc = lane & 15;
    const int ncol = wv * 16 + (lane & 15);
    const int8_t* Brow = Bw + ncol * 320;
    v8i acc = {0, 0, 0, 0, 0, 0, 0, 0};
#pragma unroll
    for (int kc = 0; kc < 5; ++kc) {
        const int k0 = kc * 64;
        v8i af, bf;
#pragma unroll
        for (int v = 0; v < 8; ++v) {
            af[v] = *(const int*)(&Alds[mloc * 320 + k0 + a_koff(v, lane)]);
            bf[v] = *(const int*)(Brow + k0 + b_koff(v, lane));
        }
        acc = __builtin_amdgcn_wmma_i32_16x16x64_iu8(true, af, true, bf, acc,
                                                     false, false);
    }
    int m03 = acc[0];
    m03 = acc[1] > m03 ? acc[1] : m03;
    m03 = acc[2] > m03 ? acc[2] : m03;
    m03 = acc[3] > m03 ? acc[3] : m03;
    int m47 = acc[4];
    m47 = acc[5] > m47 ? acc[5] : m47;
    m47 = acc[6] > m47 ? acc[6] : m47;
    m47 = acc[7] > m47 ? acc[7] : m47;
    float inv = g[ncol] / sqrtf(bv[ncol] + EPSBN);
    float beta = bb[ncol] - bm[ncol] * inv;
    float bi = bconv[ncol];
    float h0 = fminf(1.f, fmaxf(-1.f, ((float)m03 + bi) * inv + beta));
    float h1 = fminf(1.f, fmaxf(-1.f, ((float)m47 + bi) * inv + beta));
    int qlo = (lane & 16) ? 2 : 0;
    s2[b * 4096 + ncol * 64 + pl0 + qlo] = sgn8(h0);
    s2[b * 4096 + ncol * 64 + pl0 + qlo + 1] = sgn8(h1);
}

template <int K, int N, bool SIGN_OUT>
__global__ __launch_bounds__(128) void k_fc_wmma(
    const int8_t* __restrict__ A, const int8_t* __restrict__ Bm,
    const float* __restrict__ bias, const float* __restrict__ g,
    const float* __restrict__ bb, const float* __restrict__ bm,
    const float* __restrict__ bv, int8_t* __restrict__ out8,
    float* __restrict__ outf) {
    const int lane = threadIdx.x & 31;
    const int wid = blockIdx.x * 4 + (threadIdx.x >> 5);
    const int tilesN = N / 16;
    const int tileN = wid % tilesN;
    const int tileM = wid / tilesN;
    const int mrow = tileM * 16 + (lane & 15);
    const int ncol = tileN * 16 + (lane & 15);
    const int8_t* Arow = A + (size_t)mrow * K;
    const int8_t* Brow = Bm + (size_t)ncol * K;
    v8i acc = {0, 0, 0, 0, 0, 0, 0, 0};
    for (int kc = 0; kc < K / 64; ++kc) {
        const int k0 = kc * 64;
        if (kc + 1 < K / 64) {
            __builtin_prefetch(Arow + k0 + 64, 0, 0);
            __builtin_prefetch(Brow + k0 + 64, 0, 0);
        }
        v8i af, bf;
#pragma unroll
        for (int v = 0; v < 8; ++v) {
            af[v] = *(const int*)(Arow + k0 + a_koff(v, lane));
            bf[v] = *(const int*)(Brow + k0 + b_koff(v, lane));
        }
        acc = __builtin_amdgcn_wmma_i32_16x16x64_iu8(true, af, true, bf, acc,
                                                     false, false);
    }
    float inv = g[ncol] / sqrtf(bv[ncol] + EPSBN);
    float beta = bb[ncol] - bm[ncol] * inv;
    float bi = bias[ncol];
    const int mbase = tileM * 16 + ((lane & 16) ? 8 : 0);
#pragma unroll
    for (int j = 0; j < 8; ++j) {
        float h = (float)acc[j] + bi;
        h = h * inv + beta;
        h = fminf(1.f, fmaxf(-1.f, h));
        if (SIGN_OUT)
            out8[(size_t)(mbase + j) * N + ncol] = sgn8(h);
        else
            outf[(size_t)(mbase + j) * N + ncol] = h;
    }
}

__global__ __launch_bounds__(256) void k_fc3(const float* __restrict__ X,
                                             const float* __restrict__ W,
                                             const float* __restrict__ b,
                                             float* __restrict__ out) {
    const int lane = threadIdx.x & 31;
    const int row = blockIdx.x * 8 + (threadIdx.x >> 5);
    float acc[10];
#pragma unroll
    for (int j = 0; j < 10; ++j) acc[j] = 0.f;
    for (int k = lane; k < 256; k += 32) {
        float xv = X[row * 256 + k];
#pragma unroll
        for (int j = 0; j < 10; ++j) acc[j] += xv * W[j * 256 + k];
    }
#pragma unroll
    for (int j = 0; j < 10; ++j) {
#pragma unroll
        for (int m = 16; m >= 1; m >>= 1) acc[j] += __shfl_xor(acc[j], m, 32);
        acc[j] += b[j];
    }
    float mx = acc[0];
#pragma unroll
    for (int j = 1; j < 10; ++j) mx = fmaxf(mx, acc[j]);
    float se = 0.f;
#pragma unroll
    for (int j = 0; j < 10; ++j) se += expf(acc[j] - mx);
    float lse = mx + logf(se);
    if (lane < 10) out[row * 10 + lane] = acc[lane] - lse;
}

extern "C" void kernel_launch(void* const* d_in, const int* in_sizes, int n_in,
                              void* d_out, int out_size, void* d_ws,
                              size_t ws_size, hipStream_t stream) {
    const float* x       = (const float*)d_in[0];
    const float* conv1_w = (const float*)d_in[1];
    const float* conv1_b = (const float*)d_in[2];
    const float* bn1_g   = (const float*)d_in[3];
    const float* bn1_b   = (const float*)d_in[4];
    const float* bn1_m   = (const float*)d_in[5];
    const float* bn1_v   = (const float*)d_in[6];
    const float* conv2_w = (const float*)d_in[7];
    const float* conv2_b = (const float*)d_in[8];
    const float* bn2_g   = (const float*)d_in[9];
    const float* bn2_b   = (const float*)d_in[10];
    const float* bn2_m   = (const float*)d_in[11];
    const float* bn2_v   = (const float*)d_in[12];
    const float* fc1_w   = (const float*)d_in[13];
    const float* fc1_b   = (const float*)d_in[14];
    const float* bn3_g   = (const float*)d_in[15];
    const float* bn3_b   = (const float*)d_in[16];
    const float* bn3_m   = (const float*)d_in[17];
    const float* bn3_v   = (const float*)d_in[18];
    const float* fc2_w   = (const float*)d_in[19];
    const float* fc2_b   = (const float*)d_in[20];
    const float* bn4_g   = (const float*)d_in[21];
    const float* bn4_b   = (const float*)d_in[22];
    const float* bn4_m   = (const float*)d_in[23];
    const float* bn4_v   = (const float*)d_in[24];
    const float* fc3_w   = (const float*)d_in[25];
    const float* fc3_b   = (const float*)d_in[26];
    (void)in_sizes; (void)n_in; (void)out_size; (void)ws_size;

    char* ws = (char*)d_ws;
    int8_t* s1   = (int8_t*)(ws);
    int8_t* s2   = (int8_t*)(ws + 16777216);
    int8_t* s3   = (int8_t*)(ws + 25165824);
    float*  s4   = (float*)(ws + 26214400);
    int8_t* Bw2  = (int8_t*)(ws + 28311552);
    int8_t* Bfc1 = (int8_t*)(ws + 28332032);
    int8_t* Bfc2 = (int8_t*)(ws + 30429184);

    k_prep<<<2048, 256, 0, stream>>>(conv2_w, fc1_w, fc2_w, Bw2, Bfc1, Bfc2);
    k_conv1<<<2048 * 32, 256, 0, stream>>>(x, conv1_w, conv1_b, bn1_g, bn1_b,
                                           bn1_m, bn1_v, s1);
    k_conv2<<<2048 * 16, 128, 0, stream>>>(s1, Bw2, conv2_b, bn2_g, bn2_b,
                                           bn2_m, bn2_v, s2);
    k_fc_wmma<4096, 512, true><<<1024, 128, 0, stream>>>(
        s2, Bfc1, fc1_b, bn3_g, bn3_b, bn3_m, bn3_v, s3, nullptr);
    k_fc_wmma<512, 256, false><<<512, 128, 0, stream>>>(
        s3, Bfc2, fc2_b, bn4_g, bn4_b, bn4_m, bn4_v, nullptr, s4);
    k_fc3<<<256, 256, 0, stream>>>(s4, fc3_w, fc3_b, (float*)d_out);
}
